// MultiHeadAttention_56066503082010
// MI455X (gfx1250) — matrix operand order repaired and checked by reading, compiled, not run on hardware
//
#include <hip/hip_runtime.h>
#include <hip/hip_bf16.h>

#define DEV __device__ __forceinline__

typedef __bf16 bf16_t;
typedef __attribute__((ext_vector_type(16))) __bf16 v16bf;
typedef __attribute__((ext_vector_type(8)))  __bf16 v8bf;
typedef __attribute__((ext_vector_type(8)))  float  v8f;

typedef unsigned int u32x4 __attribute__((ext_vector_type(4)));
typedef int          i32x8 __attribute__((ext_vector_type(8)));
typedef int          i32x4 __attribute__((ext_vector_type(4)));

#if __has_builtin(__builtin_amdgcn_tensor_load_to_lds) && \
    __has_builtin(__builtin_amdgcn_s_wait_tensorcnt)
#define USE_TDM 1
#else
#define USE_TDM 0
#endif

constexpr int B = 4, T = 2048, D = 1024, H = 16, HD = 64;

// ---------------------------------------------------------------------------
// WMMA fragment loaders (bf16, wave32). Layouts per CDNA5 ISA 7.12.2:
//  A (16x32, MxK): lanes 0-15: M=lane, K = 0..7 then 16..23
//                  lanes 16-31: M=lane-16, K = 8..15 then 24..31
//  B (32x16, KxN) stored as row-major N x K ("B transposed"):
//                  lanes 0-15: N=lane, K = 0..7 then 16..23
//                  lanes 16-31: N=lane-16, K = 8..15 then 24..31
// ---------------------------------------------------------------------------
DEV v16bf load_a_frag(const bf16_t* tile, int ld) {
  const int lane = threadIdx.x & 31;
  const int row  = lane & 15;
  const int koff = (lane & 16) ? 8 : 0;
  const bf16_t* p = tile + (size_t)row * ld + koff;
  v8bf lo = *(const v8bf*)(p);
  v8bf hi = *(const v8bf*)(p + 16);
  v16bf r;
#pragma unroll
  for (int i = 0; i < 8; ++i) { r[i] = lo[i]; r[i + 8] = hi[i]; }
  return r;
}

DEV v16bf load_b_frag(const bf16_t* tileT, int ld) {
  const int lane = threadIdx.x & 31;
  const int n  = lane & 15;
  const int kb = (lane & 16) ? 8 : 0;
  const bf16_t* p = tileT + (size_t)n * ld + kb;
  v8bf lo = *(const v8bf*)(p);
  v8bf hi = *(const v8bf*)(p + 16);
  v16bf r;
#pragma unroll
  for (int i = 0; i < 8; ++i) { r[i] = lo[i]; r[i + 8] = hi[i]; }
  return r;
}

DEV v8f wmma_bf16(v16bf a, v16bf b, v8f c) {
  return __builtin_amdgcn_wmma_f32_16x16x32_bf16(false, a, false, b,
                                                 (short)0, c, false, false);
}

// C/D 16x16 f32 layout: lane&15 = N; element e -> M = e + ((lane&16)?8:0)
DEV void store_tile_bf16(bf16_t* out, int ld, v8f acc) {
  const int lane = threadIdx.x & 31;
  const int n  = lane & 15;
  const int mb = (lane & 16) ? 8 : 0;
#pragma unroll
  for (int e = 0; e < 8; ++e) out[(size_t)(mb + e) * ld + n] = (bf16_t)acc[e];
}

#if USE_TDM
// ---------------------------------------------------------------------------
// Tensor Data Mover: stage a 2D bf16 tile (tile_w x tile_h, row stride in
// elements) from global memory into LDS. D# layout per CDNA5 ISA 08 §8.
// One instruction per wave; tracked with TENSORcnt. 6-arg builtin form.
// ---------------------------------------------------------------------------
DEV void tdm_load_2d_bf16(const bf16_t* gsrc, const bf16_t* lds_dst,
                          int tile_w, int tile_h, int row_stride) {
  const unsigned long long ga = (unsigned long long)gsrc;
  // Flat shared-pointer low 32 bits == LDS byte address (aperture truncation).
  const unsigned lds_addr = (unsigned)(unsigned long long)lds_dst;

  u32x4 g0;
  g0[0] = 1u;                                       // count=1, user descriptor
  g0[1] = lds_addr;                                 // lds_addr [63:32]
  g0[2] = (unsigned)(ga & 0xFFFFFFFFu);             // global_addr [95:64]
  g0[3] = (unsigned)((ga >> 32) & 0x01FFFFFFu)      // global_addr [120:96]
        | (2u << 30);                               // type=2 ("image") [127:126]

  const unsigned td0 = 1u << 20, td1 = 1u << 20;    // generous tensor extents
  i32x8 g1;
  g1[0] = (int)(1u << 16);                          // data_size=2B; mask/flags=0
  g1[1] = (int)((td0 & 0xFFFFu) << 16);             // tensor_dim0 [15:0]
  g1[2] = (int)(((td0 >> 16) & 0xFFFFu) |           // tensor_dim0 [31:16]
                ((td1 & 0xFFFFu) << 16));           // tensor_dim1 [15:0]
  g1[3] = (int)(((td1 >> 16) & 0xFFFFu) |           // tensor_dim1 [31:16]
                ((unsigned)tile_w << 16));          // tile_dim0
  g1[4] = (int)((unsigned)tile_h & 0xFFFFu);        // tile_dim1; tile_dim2=0
  g1[5] = (int)(unsigned)row_stride;                // tensor_dim0_stride lo
  g1[6] = 0;                                        // stride hi / dim1_stride lo
  g1[7] = 0;                                        // dim1_stride hi

  const i32x4 z4 = {0, 0, 0, 0};                    // groups 2/3 unused (2D)
  const i32x8 z8 = {0, 0, 0, 0, 0, 0, 0, 0};        // extra group (unused)
  __builtin_amdgcn_tensor_load_to_lds(g0, g1, z4, z4, z8, 0);
}
#endif

// ---------------------------------------------------------------------------
// Prep kernels
// ---------------------------------------------------------------------------
__global__ void k_cast(const float* __restrict__ in, bf16_t* __restrict__ out,
                       size_t n) {
  size_t i = (size_t)blockIdx.x * blockDim.x + threadIdx.x;
  if (i < n) out[i] = (bf16_t)in[i];
}

// Wq/Wk/Wv: [H][D][HD] f32 -> Wqkvt: [3][H][HD][D] bf16 (transposed per head)
__global__ void k_prep_w(const float* __restrict__ Wq,
                         const float* __restrict__ Wk,
                         const float* __restrict__ Wv,
                         bf16_t* __restrict__ Wqkvt) {
  size_t i = (size_t)blockIdx.x * blockDim.x + threadIdx.x;
  const size_t per = (size_t)H * D * HD;
  if (i >= 3 * per) return;
  const int z = (int)(i / per);
  size_t r = i % per;
  const int h = (int)(r / ((size_t)D * HD));
  size_t r2 = r % ((size_t)D * HD);
  const int d = (int)(r2 / HD);
  const int n = (int)(r2 % HD);
  const float* W = (z == 0) ? Wq : (z == 1) ? Wk : Wv;
  Wqkvt[(((size_t)z * H + h) * HD + n) * D + d] =
      (bf16_t)W[((size_t)h * D + d) * HD + n];
}

// ---------------------------------------------------------------------------
// QKV projection: one wave computes a 64x64 tile of Q, K, or V for one head.
// 16 WMMAs per k-step. Unroll-by-2 ping-pong register/LDS buffers: no
// rotation moves, no WMMA->VALU hazard NOPs.
// z==2 (V) is stored transposed: Vt[b][h][hd][t].
// ---------------------------------------------------------------------------
__global__ void __launch_bounds__(32)
k_qkv(const bf16_t* __restrict__ xb, const bf16_t* __restrict__ Wqkvt,
      bf16_t* __restrict__ Q, bf16_t* __restrict__ K, bf16_t* __restrict__ Vt) {
  const int rt = blockIdx.x;                 // 0 .. B*T/64-1 (global row tile)
  const int h  = blockIdx.y;
  const int z  = blockIdx.z;                 // 0=Q 1=K 2=V
  const int b  = rt / (T / 64);
  const int t0 = (rt % (T / 64)) * 64;

  const bf16_t* A  = xb + (size_t)rt * 64 * D;
  const bf16_t* Wt = Wqkvt + ((size_t)z * H + h) * HD * D;

  v8f acc[4][4];
#pragma unroll
  for (int mt = 0; mt < 4; ++mt)
#pragma unroll
    for (int nt = 0; nt < 4; ++nt)
#pragma unroll
      for (int e = 0; e < 8; ++e) acc[mt][nt][e] = 0.0f;

  auto loadA4 = [&](v16bf (&dst)[4], int k0) {
#pragma unroll
    for (int mt = 0; mt < 4; ++mt)
      dst[mt] = load_a_frag(A + (size_t)mt * 16 * D + k0, D);
  };
  auto mma16 = [&](const v16bf (&a4)[4], const v16bf (&b4)[4]) {
#pragma unroll
    for (int mt = 0; mt < 4; ++mt)
#pragma unroll
      for (int nt = 0; nt < 4; ++nt)
        acc[mt][nt] = wmma_bf16(a4[mt], b4[nt], acc[mt][nt]);
  };

  v16bf afA[4], afB[4];
  loadA4(afA, 0);

#if USE_TDM
  __shared__ __align__(16) bf16_t wtile[2][64 * 32];
  tdm_load_2d_bf16(Wt, &wtile[0][0], 32, 64, D);
  for (int k0 = 0; k0 < D; k0 += 64) {
    // ---- phase 0: consume buffer 0 / afA, prefetch k0+32 into 1 / afB ----
    if (k0 + 32 < D) {
      loadA4(afB, k0 + 32);
      tdm_load_2d_bf16(Wt + k0 + 32, &wtile[1][0], 32, 64, D);
      __builtin_amdgcn_s_wait_tensorcnt(1);
    } else {
      __builtin_amdgcn_s_wait_tensorcnt(0);
    }
    __syncthreads();
    {
      v16bf b4[4];
#pragma unroll
      for (int nt = 0; nt < 4; ++nt)
        b4[nt] = load_b_frag(&wtile[0][nt * 16 * 32], 32);
      mma16(afA, b4);
    }
    // ---- phase 1: consume buffer 1 / afB, prefetch k0+64 into 0 / afA ----
    if (k0 + 32 < D) {
      if (k0 + 64 < D) {
        loadA4(afA, k0 + 64);
        tdm_load_2d_bf16(Wt + k0 + 64, &wtile[0][0], 32, 64, D);
        __builtin_amdgcn_s_wait_tensorcnt(1);
      } else {
        __builtin_amdgcn_s_wait_tensorcnt(0);
      }
      __syncthreads();
      v16bf b4[4];
#pragma unroll
      for (int nt = 0; nt < 4; ++nt)
        b4[nt] = load_b_frag(&wtile[1][nt * 16 * 32], 32);
      mma16(afB, b4);
    }
  }
#else
  auto loadB4 = [&](v16bf (&dst)[4], int k0) {
#pragma unroll
    for (int nt = 0; nt < 4; ++nt)
      dst[nt] = load_b_frag(Wt + (size_t)nt * 16 * D + k0, D);
  };
  v16bf bfA[4], bfB[4];
  loadB4(bfA, 0);
  for (int k0 = 0; k0 < D; k0 += 64) {
    if (k0 + 32 < D) { loadA4(afB, k0 + 32); loadB4(bfB, k0 + 32); }
    mma16(afA, bfA);
    if (k0 + 32 < D) {
      if (k0 + 64 < D) { loadA4(afA, k0 + 64); loadB4(bfA, k0 + 64); }
      mma16(afB, bfB);
    }
  }
#endif

  const size_t bh = (size_t)b * H + h;
  if (z == 2) {
    bf16_t* vbase = Vt + bh * HD * T;
    const int lane = threadIdx.x & 31;
    const int n  = lane & 15;
    const int mb = (lane & 16) ? 8 : 0;
#pragma unroll
    for (int mt = 0; mt < 4; ++mt)
#pragma unroll
      for (int nt = 0; nt < 4; ++nt)
#pragma unroll
        for (int e = 0; e < 8; ++e)
          vbase[(size_t)(nt * 16 + n) * T + (t0 + mt * 16 + mb + e)] =
              (bf16_t)acc[mt][nt][e];
  } else {
    bf16_t* obase = ((z == 0) ? Q : K) + (bh * T + t0) * HD;
#pragma unroll
    for (int mt = 0; mt < 4; ++mt)
#pragma unroll
      for (int nt = 0; nt < 4; ++nt)
        store_tile_bf16(obase + (size_t)mt * 16 * HD + nt * 16, HD, acc[mt][nt]);
  }
}

// ---------------------------------------------------------------------------
// Flash attention: one wave per 16-row q-tile of one (b,h). Online softmax
// over 32-column s-tiles. K fragments ping-pong between two register sets
// (no rotation copies); next K tile + current V tile are loaded before the
// softmax VALU section so memory latency hides under the reductions.
// ---------------------------------------------------------------------------
__global__ void __launch_bounds__(32)
k_attn(const bf16_t* __restrict__ Q, const bf16_t* __restrict__ K,
       const bf16_t* __restrict__ Vt, bf16_t* __restrict__ O) {
  __shared__ __align__(16) bf16_t pbuf[16 * 32];

  const int t0 = blockIdx.x * 16;
  const int h  = blockIdx.y;
  const int b  = blockIdx.z;
  const size_t bh = (size_t)b * H + h;
  const bf16_t* Qb = Q  + (bh * T + t0) * HD;
  const bf16_t* Kb = K  + bh * T * HD;
  const bf16_t* Vb = Vt + bh * HD * T;

  const int lane = threadIdx.x & 31;
  const int nIdx = lane & 15;
  const int mb   = (lane & 16) ? 8 : 0;

  const v16bf aq0 = load_a_frag(Qb, HD);
  const v16bf aq1 = load_a_frag(Qb + 32, HD);

  v8f acc[4];
  float mrow[8], lrow[8];
#pragma unroll
  for (int e = 0; e < 8; ++e) { mrow[e] = -1e30f; lrow[e] = 0.0f; }
#pragma unroll
  for (int nt = 0; nt < 4; ++nt)
#pragma unroll
    for (int e = 0; e < 8; ++e) acc[nt][e] = 0.0f;

  const int send = t0 + 16;                     // need all cols <= t0+15

  auto loadK4 = [&](v16bf (&dst)[4], int s0) {
    const bf16_t* Kp = Kb + (size_t)s0 * HD;
    dst[0] = load_b_frag(Kp, HD);
    dst[1] = load_b_frag(Kp + 32, HD);
    dst[2] = load_b_frag(Kp + (size_t)16 * HD, HD);
    dst[3] = load_b_frag(Kp + (size_t)16 * HD + 32, HD);
  };

  // Process one 32-column s-tile using kc; prefetch the next tile into knx.
  auto attn_tile = [&](int s0, const v16bf (&kc)[4], v16bf (&knx)[4]) {
    // ---- scores: two 16x16 tiles over columns [s0, s0+32) -------------
    v8f S0, S1;
#pragma unroll
    for (int e = 0; e < 8; ++e) { S0[e] = 0.0f; S1[e] = 0.0f; }
    S0 = wmma_bf16(aq0, kc[0], S0);
    S0 = wmma_bf16(aq1, kc[1], S0);
    S1 = wmma_bf16(aq0, kc[2], S1);
    S1 = wmma_bf16(aq1, kc[3], S1);

    // ---- issue V loads (current tile) and K loads (next tile) --------
    v16bf vf[4];
#pragma unroll
    for (int nt = 0; nt < 4; ++nt)
      vf[nt] = load_b_frag(Vb + (size_t)(nt * 16) * T + s0, T);
    if (s0 + 32 < send) loadK4(knx, s0 + 32);

    // ---- scale + causal mask + per-row tile max -----------------------
    float tmax[8];
#pragma unroll
    for (int e = 0; e < 8; ++e) {
      const int trow = t0 + mb + e;
      float a = S0[e] * 0.125f;                 // 1/sqrt(64)
      float c = S1[e] * 0.125f;
      if (s0 + nIdx > trow)      a = -1e30f;
      if (s0 + 16 + nIdx > trow) c = -1e30f;
      S0[e] = a; S1[e] = c;
      tmax[e] = fmaxf(a, c);
    }
#pragma unroll
    for (int e = 0; e < 8; ++e)
#pragma unroll
      for (int off = 1; off < 16; off <<= 1)
        tmax[e] = fmaxf(tmax[e], __shfl_xor(tmax[e], off, 16));
    // ---- online softmax update ---------------------------------------
    float fac[8], rsum[8];
#pragma unroll
    for (int e = 0; e < 8; ++e) {
      const float mn = fmaxf(mrow[e], tmax[e]);
      fac[e]  = __expf(mrow[e] - mn);
      mrow[e] = mn;
      const float p0 = __expf(S0[e] - mn);
      const float p1 = __expf(S1[e] - mn);
      S0[e] = p0; S1[e] = p1;
      rsum[e] = p0 + p1;
    }
#pragma unroll
    for (int e = 0; e < 8; ++e)
#pragma unroll
      for (int off = 1; off < 16; off <<= 1)
        rsum[e] += __shfl_xor(rsum[e], off, 16);
#pragma unroll
    for (int e = 0; e < 8; ++e) lrow[e] = lrow[e] * fac[e] + rsum[e];
#pragma unroll
    for (int nt = 0; nt < 4; ++nt)
#pragma unroll
      for (int e = 0; e < 8; ++e) acc[nt][e] *= fac[e];

    // ---- P: C-layout -> row-major 16x32 in LDS -> A fragment ----------
    __syncthreads();
#pragma unroll
    for (int e = 0; e < 8; ++e) {
      pbuf[(mb + e) * 32 + nIdx]      = (bf16_t)S0[e];
      pbuf[(mb + e) * 32 + 16 + nIdx] = (bf16_t)S1[e];
    }
    __syncthreads();
    const v16bf ap = load_a_frag(pbuf, 32);
    // ---- O += P @ V ----------------------------------------------------
#pragma unroll
    for (int nt = 0; nt < 4; ++nt) acc[nt] = wmma_bf16(ap, vf[nt], acc[nt]);
  };

  v16bf kA[4], kB[4];
  loadK4(kA, 0);
  for (int s0 = 0; s0 < send; s0 += 64) {
    attn_tile(s0, kA, kB);                    // prefetches s0+32 into kB
    if (s0 + 32 < send) attn_tile(s0 + 32, kB, kA);  // prefetches into kA
  }

  float rinv[8];
#pragma unroll
  for (int e = 0; e < 8; ++e) rinv[e] = 1.0f / lrow[e];
  bf16_t* Ob = O + ((size_t)b * T + t0) * D + h * HD;
#pragma unroll
  for (int nt = 0; nt < 4; ++nt)
#pragma unroll
    for (int e = 0; e < 8; ++e)
      Ob[(size_t)(mb + e) * D + nt * 16 + nIdx] = (bf16_t)(acc[nt][e] * rinv[e]);
}

// ---------------------------------------------------------------------------
// Output projection: out = O @ Wo^T + bo. 64x64 tile per wave, unroll-by-2
// ping-pong; weight tiles staged into LDS via TDM when available.
// ---------------------------------------------------------------------------
__global__ void __launch_bounds__(32)
k_oproj(const bf16_t* __restrict__ O, const bf16_t* __restrict__ Wob,
        const float* __restrict__ bo, float* __restrict__ out) {
  const int rt = blockIdx.x;                   // 0 .. B*T/64-1
  const int n0 = blockIdx.y * 64;
  const bf16_t* A  = O + (size_t)rt * 64 * D;
  const bf16_t* Bw = Wob + (size_t)n0 * D;

  v8f acc[4][4];
#pragma unroll
  for (int mt = 0; mt < 4; ++mt)
#pragma unroll
    for (int nt = 0; nt < 4; ++nt)
#pragma unroll
      for (int e = 0; e < 8; ++e) acc[mt][nt][e] = 0.0f;

  auto loadA4 = [&](v16bf (&dst)[4], int k0) {
#pragma unroll
    for (int mt = 0; mt < 4; ++mt)
      dst[mt] = load_a_frag(A + (size_t)mt * 16 * D + k0, D);
  };
  auto mma16 = [&](const v16bf (&a4)[4], const v16bf (&b4)[4]) {
#pragma unroll
    for (int mt = 0; mt < 4; ++mt)
#pragma unroll
      for (int nt = 0; nt < 4; ++nt)
        acc[mt][nt] = wmma_bf16(a4[mt], b4[nt], acc[mt][nt]);
  };

  v16bf afA[4], afB[4];
  loadA4(afA, 0);

#if USE_TDM
  __shared__ __align__(16) bf16_t btile[2][64 * 32];
  tdm_load_2d_bf16(Bw, &btile[0][0], 32, 64, D);
  for (int k0 = 0; k0 < D; k0 += 64) {
    if (k0 + 32 < D) {
      loadA4(afB, k0 + 32);
      tdm_load_2d_bf16(Bw + k0 + 32, &btile[1][0], 32, 64, D);
      __builtin_amdgcn_s_wait_tensorcnt(1);
    } else {
      __builtin_amdgcn_s_wait_tensorcnt(0);
    }
    __syncthreads();
    {
      v16bf b4[4];
#pragma unroll
      for (int nt = 0; nt < 4; ++nt)
        b4[nt] = load_b_frag(&btile[0][nt * 16 * 32], 32);
      mma16(afA, b4);
    }
    if (k0 + 32 < D) {
      if (k0 + 64 < D) {
        loadA4(afA, k0 + 64);
        tdm_load_2d_bf16(Bw + k0 + 64, &btile[0][0], 32, 64, D);
        __builtin_amdgcn_s_wait_tensorcnt(1);
      } else {
        __builtin_amdgcn_s_wait_tensorcnt(0);
      }
      __syncthreads();
      v16bf b4[4];
#pragma unroll
      for (int nt = 0; nt < 4; ++nt)
        b4[nt] = load_b_frag(&btile[1][nt * 16 * 32], 32);
      mma16(afB, b4);
    }
  }
#else
  auto loadB4 = [&](v16bf (&dst)[4], int k0) {
#pragma unroll
    for (int nt = 0; nt < 4; ++nt)
      dst[nt] = load_b_frag(Bw + (size_t)nt * 16 * D + k0, D);
  };
  v16bf bfA[4], bfB[4];
  loadB4(bfA, 0);
  for (int k0 = 0; k0 < D; k0 += 64) {
    if (k0 + 32 < D) { loadA4(afB, k0 + 32); loadB4(bfB, k0 + 32); }
    mma16(afA, bfA);
    if (k0 + 32 < D) {
      if (k0 + 64 < D) { loadA4(afA, k0 + 64); loadB4(bfA, k0 + 64); }
      mma16(afB, bfB);
    }
  }
#endif

  const int ln = threadIdx.x & 31;
  const int n  = ln & 15;
  const int mbv = (ln & 16) ? 8 : 0;
#pragma unroll
  for (int nt = 0; nt < 4; ++nt) {
    const float bias = bo[n0 + nt * 16 + n];
#pragma unroll
    for (int mt = 0; mt < 4; ++mt)
#pragma unroll
      for (int e = 0; e < 8; ++e)
        out[((size_t)rt * 64 + mt * 16 + mbv + e) * D + n0 + nt * 16 + n] =
            acc[mt][nt][e] + bias;
  }
}

// ---------------------------------------------------------------------------
// Host launch. Workspace layout (bf16 elements), total ~88 MB:
//   xb[B*T*D] | Wqkvt[3*H*HD*D] | Wob[D*D] | Q | K | Vt (B*H*T*HD each) | O[B*T*D]
// ---------------------------------------------------------------------------
extern "C" void kernel_launch(void* const* d_in, const int* in_sizes, int n_in,
                              void* d_out, int out_size, void* d_ws, size_t ws_size,
                              hipStream_t stream) {
  (void)in_sizes; (void)n_in; (void)out_size; (void)ws_size;
  const float* x  = (const float*)d_in[0];
  const float* Wq = (const float*)d_in[1];
  const float* Wk = (const float*)d_in[2];
  const float* Wv = (const float*)d_in[3];
  const float* Wo = (const float*)d_in[4];
  const float* bo = (const float*)d_in[5];
  float* out = (float*)d_out;

  const size_t N_X    = (size_t)B * T * D;        // 8M
  const size_t N_WQKV = (size_t)3 * H * HD * D;   // 3M
  const size_t N_WO   = (size_t)D * D;            // 1M
  const size_t N_HEAD = (size_t)B * H * T * HD;   // 8M

  bf16_t* xb    = (bf16_t*)d_ws;
  bf16_t* Wqkvt = xb + N_X;
  bf16_t* Wob   = Wqkvt + N_WQKV;
  bf16_t* Qb    = Wob + N_WO;
  bf16_t* Kb    = Qb + N_HEAD;
  bf16_t* Vt    = Kb + N_HEAD;
  bf16_t* Ob    = Vt + N_HEAD;

  k_cast<<<dim3((unsigned)(N_X / 256)), 256, 0, stream>>>(x, xb, N_X);
  k_cast<<<dim3((unsigned)(N_WO / 256)), 256, 0, stream>>>(Wo, Wob, N_WO);
  k_prep_w<<<dim3((unsigned)(N_WQKV / 256)), 256, 0, stream>>>(Wq, Wk, Wv, Wqkvt);

  k_qkv<<<dim3(B * T / 64, H, 3), 32, 0, stream>>>(xb, Wqkvt, Qb, Kb, Vt);
  k_attn<<<dim3(T / 16, H, B), 32, 0, stream>>>(Qb, Kb, Vt, Ob);
  k_oproj<<<dim3(B * T / 64, D / 64), 32, 0, stream>>>(Ob, Wob, bo, out);
}
